// E2EformerBlockCore_65137474011742
// MI455X (gfx1250) — matrix operand order repaired and checked by reading, not compiled, not run on hardware
//
#include <hip/hip_runtime.h>

// ---------------------------------------------------------------------------
// Types for CDNA5 WMMA (wave32, 16x16x32 bf16 -> f32)
// ---------------------------------------------------------------------------
typedef __bf16 bf16;
typedef __attribute__((ext_vector_type(16))) __bf16 v16bf;
typedef __attribute__((ext_vector_type(8)))  __bf16 v8bf;
typedef __attribute__((ext_vector_type(4)))  __bf16 v4bf;
typedef __attribute__((ext_vector_type(8)))  float   v8f;

static_assert(sizeof(bf16) == 2, "bf16 must be 2 bytes");

__device__ __forceinline__ v16bf cat8(v8bf lo, v8bf hi) {
  return __builtin_shufflevector(lo, hi, 0,1,2,3,4,5,6,7,8,9,10,11,12,13,14,15);
}
__device__ __forceinline__ v8f wmma_bf16(v16bf a, v16bf b, v8f c) {
  return __builtin_amdgcn_wmma_f32_16x16x32_bf16(false, a, false, b, (short)0, c, false, false);
}
__device__ __forceinline__ float sigm(float x) { return 1.f / (1.f + __expf(-x)); }

// ---------------------------------------------------------------------------
// Generic WMMA GEMM: C[M,N] = epilogue(A[M,K] @ B[K,N])
//   A, B bf16.  BM=128 (8 waves x 16 rows), BN = NSUB*16, BK=32.
//   Double-buffered LDS staging + global_prefetch of tile k+2.
//   A gather modes: direct row-major, transposed (A[k*lda+row]), OPM reshape.
// ---------------------------------------------------------------------------
enum { EPI_BF16 = 0, EPI_RELU_BF16, EPI_F32, EPI_SIG_F32, EPI_MASK_BF16,
       EPI_GATE_BF16, EPI_RES_MASK_F32, EPI_RES_DIV_F32, EPI_RES_AUX_F32,
       EPI_RES_F32, EPI_SCAT_CH_F32 };
enum { AG_DIR = 0, AG_TRANS, AG_OPM2 };

struct GemmP {
  const bf16* A; const bf16* B;
  const float* bias; const float* res; const float* aux;
  const float* rowmask; const float* norm;
  void* out;
  int M, N, K, lda, ldb, ldo;
  int out_trans;                   // scatter rows transposed (row=i*256+j -> j*256+i)
  long strideA, strideB;           // batch strides (elements)
};

template <int EPI, int AG, int NSUB>
__global__ __launch_bounds__(256) void gemm_k(GemmP p) {
  constexpr int BN = NSUB * 16;
  __shared__ bf16 As[2][128][40];    // [buf][m][k], stride 40 keeps 16B alignment
  __shared__ bf16 BsT[2][BN][40];    // [buf][n][k] (B staged transposed)

  const int tid  = threadIdx.x;
  const int wave = tid >> 5;
  const int lane = tid & 31;
  const int lx   = lane & 15;
  const int half = lane >> 4;
  const int m0   = blockIdx.x * 128;
  const int n0   = blockIdx.y * BN;
  const bf16* A = p.A + (long)blockIdx.z * p.strideA;
  const bf16* B = p.B + (long)blockIdx.z * p.strideB;

  // ---- LDS staging helpers ----
  auto stageA = [&](int buf, int k0) {
    if constexpr (AG == AG_DIR) {
#pragma unroll
      for (int it = 0; it < 4; ++it) {            // 1024 quads
        int qi = it * 256 + tid;
        int m = qi >> 3, kq = (qi & 7) * 4;
        *(v4bf*)&As[buf][m][kq] = *(const v4bf*)(A + (long)(m0 + m) * p.lda + k0 + kq);
      }
    } else if constexpr (AG == AG_TRANS) {
#pragma unroll
      for (int it = 0; it < 16; ++it) {           // 4096 scalars, coalesced over m
        int e = it * 256 + tid;
        int m = e & 127, k = e >> 7;
        As[buf][m][k] = A[(long)(k0 + k) * p.lda + m0 + m];
      }
    } else {                                      // AG_OPM2: A[(di*32+c)*8192 + j*32 + d]
#pragma unroll
      for (int it = 0; it < 16; ++it) {
        int e = it * 256 + tid;
        int k = e & 31, m = e >> 5;
        int row = m0 + m, kk = k0 + k;
        int di = row >> 8, j = row & 255, c = kk >> 5, d = kk & 31;
        As[buf][m][k] = A[(long)(di * 32 + c) * 8192 + j * 32 + d];
      }
    }
  };
  auto stageB = [&](int buf, int k0) {
    constexpr int quads = BN * 32 / 4;
    constexpr int perrow = BN / 4;
    for (int qi = tid; qi < quads; qi += 256) {
      int k = qi / perrow, nq = (qi % perrow) * 4;
      const bf16* src = B + (long)(k0 + k) * p.ldb + n0 + nq;
      bf16 x0 = src[0], x1 = src[1], x2 = src[2], x3 = src[3];
      BsT[buf][nq + 0][k] = x0; BsT[buf][nq + 1][k] = x1;
      BsT[buf][nq + 2][k] = x2; BsT[buf][nq + 3][k] = x3;
    }
  };

  v8f acc[NSUB];
#pragma unroll
  for (int i = 0; i < NSUB; ++i) acc[i] = (v8f){0,0,0,0,0,0,0,0};

  const int nk = p.K >> 5;
  stageA(0, 0);
  stageB(0, 0);

  for (int kt = 0; kt < nk; ++kt) {
    __syncthreads();                               // tile kt visible; buf kt^1 free
    const int buf = kt & 1;

    // prefetch tile kt+2 into cache hierarchy (global_prefetch_b8)
    int kpf = (kt + 2) * 32;
    if (kpf < p.K) {
      if (tid < 128) {
        if constexpr (AG == AG_DIR)
          __builtin_prefetch(A + (long)(m0 + tid) * p.lda + kpf, 0, 3);
        else if constexpr (AG == AG_TRANS)
          __builtin_prefetch(A + (long)(kpf + (tid & 31)) * p.lda + m0 + (tid >> 5) * 32, 0, 3);
        else
          __builtin_prefetch(A + (long)(((m0 + tid) >> 8) * 32 + (kpf >> 5)) * 8192 +
                                 ((m0 + tid) & 255) * 32, 0, 3);
      } else if (tid < 160) {
        __builtin_prefetch(B + (long)(kpf + tid - 128) * p.ldb + n0, 0, 3);
      }
    }
    // stage next tile into the other buffer while this one computes
    if (kt + 1 < nk) { stageA(buf ^ 1, (kt + 1) * 32); stageB(buf ^ 1, (kt + 1) * 32); }

    // ---- WMMA: A frag per wave (rows wave*16..+15), NSUB B frags ----
    int mrow = wave * 16 + lx;
    v16bf afrag = cat8(*(const v8bf*)&As[buf][mrow][half * 16],
                       *(const v8bf*)&As[buf][mrow][half * 16 + 8]);
#pragma unroll
    for (int nt = 0; nt < NSUB; ++nt) {
      int ncol = nt * 16 + lx;
      v16bf bfrag = cat8(*(const v8bf*)&BsT[buf][ncol][half * 16],
                         *(const v8bf*)&BsT[buf][ncol][half * 16 + 8]);
      acc[nt] = wmma_bf16(afrag, bfrag, acc[nt]);
    }
  }

  // ---- epilogue ----
#pragma unroll
  for (int nt = 0; nt < NSUB; ++nt) {
    int col = n0 + nt * 16 + lx;
    float bsc = p.bias ? p.bias[col] : 0.f;
#pragma unroll
    for (int r = 0; r < 8; ++r) {
      int row = m0 + wave * 16 + half * 8 + r;
      float v = acc[nt][r] + bsc;
      long orow = p.out_trans ? (long)((row & 255) * 256 + (row >> 8)) : (long)row;
      long idx = orow * (long)p.ldo + col;
      if constexpr (EPI == EPI_BF16)          ((bf16*)p.out)[idx] = (bf16)v;
      else if constexpr (EPI == EPI_RELU_BF16)((bf16*)p.out)[idx] = (bf16)fmaxf(v, 0.f);
      else if constexpr (EPI == EPI_F32)      ((float*)p.out)[idx] = v;
      else if constexpr (EPI == EPI_SIG_F32)  ((float*)p.out)[idx] = sigm(v);
      else if constexpr (EPI == EPI_MASK_BF16)((bf16*)p.out)[idx] = (bf16)(v * p.rowmask[row]);
      else if constexpr (EPI == EPI_GATE_BF16)((bf16*)p.out)[idx] = (bf16)(sigm(v) * p.aux[idx] * p.rowmask[row]);
      else if constexpr (EPI == EPI_RES_MASK_F32) ((float*)p.out)[idx] = p.res[idx] + v * p.rowmask[row];
      else if constexpr (EPI == EPI_RES_DIV_F32)  ((float*)p.out)[idx] += v / (1e-3f + p.norm[row]);
      else if constexpr (EPI == EPI_RES_AUX_F32)  ((float*)p.out)[idx] += v * p.aux[idx];
      else if constexpr (EPI == EPI_RES_F32)      ((float*)p.out)[idx] += v;
      else if constexpr (EPI == EPI_SCAT_CH_F32)
        ((float*)p.out)[((long)row * 256 + col) * 128 + blockIdx.z] = v;
    }
  }
}

// ---------------------------------------------------------------------------
// LayerNorm (f32 in, bf16 out). One wave per row. Optional transposed gather
// (in row = j*256+i when working on z^T).
// ---------------------------------------------------------------------------
__global__ __launch_bounds__(256) void ln_k(const float* __restrict__ in,
                                            bf16* __restrict__ out,
                                            const float* __restrict__ g,
                                            const float* __restrict__ b,
                                            int W, int rows, int trans) {
  int wave = threadIdx.x >> 5, lane = threadIdx.x & 31;
  int row = blockIdx.x * 8 + wave;
  if (row >= rows) return;
  long irow = trans ? (long)((row & 255) * 256 + (row >> 8)) : (long)row;
  const float* x = in + irow * (long)W;
  float xs[8];
  int nper = W >> 5;
  float s = 0.f;
  for (int i = 0; i < nper; ++i) { xs[i] = x[lane + i * 32]; s += xs[i]; }
  for (int o = 16; o >= 1; o >>= 1) s += __shfl_xor(s, o);
  float mean = s / (float)W;
  float v = 0.f;
  for (int i = 0; i < nper; ++i) { float d = xs[i] - mean; v += d * d; }
  for (int o = 16; o >= 1; o >>= 1) v += __shfl_xor(v, o);
  float rstd = rsqrtf(v / (float)W + 1e-5f);
  for (int i = 0; i < nper; ++i) {
    int c = lane + i * 32;
    out[(long)row * W + c] = (bf16)((xs[i] - mean) * rstd * g[c] + b[c]);
  }
}

// f32 -> bf16 weight convert
__global__ void f2bf_k(const float* __restrict__ in, bf16* __restrict__ out, int n) {
  int t = blockIdx.x * 256 + threadIdx.x;
  if (t < n) out[t] = (bf16)in[t];
}

// dst[c][r0][r1] = src[(swap ? r1*256+r0 : r0*256+r1)*128 + c]   (tri-mul operand prep)
__global__ void trans_cik_k(const bf16* __restrict__ src, bf16* __restrict__ dst, int swap) {
  long t = (long)blockIdx.x * 256 + threadIdx.x;     // t = c<<16 | r0<<8 | r1
  int r1 = (int)(t & 255), r0 = (int)((t >> 8) & 255), c = (int)(t >> 16);
  long sidx = swap ? ((long)(r1 * 256 + r0) * 128 + c)
                   : ((long)(r0 * 256 + r1) * 128 + c);
  dst[t] = src[sidx];
}

// norm[i*256+j] = sum_s mask[s,i]*mask[s,j]
__global__ void opmnorm_k(const float* __restrict__ msk, float* __restrict__ norm) {
  int t = blockIdx.x * 256 + threadIdx.x;
  int i = t >> 8, j = t & 255;
  float s = 0.f;
  for (int si = 0; si < 128; ++si) s += msk[si * 256 + i] * msk[si * 256 + j];
  norm[t] = s;
}

// tb bias: tbb[h*65536 + row] = sum_c ln[row,c] * wtb[c,h]
__global__ void tb_k(const bf16* __restrict__ lnz, const float* __restrict__ wtb,
                     float* __restrict__ tbb) {
  int row = blockIdx.x * 256 + threadIdx.x;
  float a0 = 0, a1 = 0, a2 = 0, a3 = 0;
  const bf16* x = lnz + (long)row * 128;
  for (int c = 0; c < 128; ++c) {
    float xv = (float)x[c];
    a0 += xv * wtb[c * 4 + 0]; a1 += xv * wtb[c * 4 + 1];
    a2 += xv * wtb[c * 4 + 2]; a3 += xv * wtb[c * 4 + 3];
  }
  tbb[0 * 65536 + row] = a0; tbb[1 * 65536 + row] = a1;
  tbb[2 * 65536 + row] = a2; tbb[3 * 65536 + row] = a3;
}

// ---------------------------------------------------------------------------
// Fused triangle attention: one workgroup = (i, h, 64-row q block), 4 waves.
// Full 256-key logit row-block in VGPRs, in-register softmax, WMMA for QK^T
// and attn@V (attn tile converted C->A layout via per-wave LDS patch).
// obuf[row,h*32+c] = (attn@V / rowsum) * sigmoid_gate
// ---------------------------------------------------------------------------
__global__ __launch_bounds__(128) void attn_k(const bf16* __restrict__ qb,
                                              const bf16* __restrict__ kb,
                                              const bf16* __restrict__ vb,
                                              const float* __restrict__ gb,
                                              const float* __restrict__ tbb,
                                              const float* __restrict__ pmask,
                                              bf16* __restrict__ ob,
                                              int mask_trans) {
  __shared__ bf16 Qs[64][40];
  __shared__ bf16 Ks[256][40];
  __shared__ bf16 VsT[32][264];      // transposed: [c][k]
  __shared__ bf16 Aw[4][16][40];     // per-wave attn tile staging (16x32)

  const int i = blockIdx.z, h = blockIdx.y, q0 = blockIdx.x * 64;
  const int tid = threadIdx.x, wave = tid >> 5, lane = tid & 31;
  const int lx = lane & 15, half = lane >> 4;
  const long base = (long)i * 256 * 128 + h * 32;

#pragma unroll
  for (int it = 0; it < 4; ++it) {                 // Q: 64x32
    int qi = it * 128 + tid; int r = qi >> 3, cq = (qi & 7) * 4;
    *(v4bf*)&Qs[r][cq] = *(const v4bf*)(qb + base + (long)(q0 + r) * 128 + cq);
  }
#pragma unroll
  for (int it = 0; it < 16; ++it) {                // K: 256x32
    int qi = it * 128 + tid; int r = qi >> 3, cq = (qi & 7) * 4;
    *(v4bf*)&Ks[r][cq] = *(const v4bf*)(kb + base + (long)r * 128 + cq);
  }
#pragma unroll
  for (int it = 0; it < 16; ++it) {                // V transposed
    int qi = it * 128 + tid; int r = qi >> 3, cq = (qi & 7) * 4;
    v4bf v = *(const v4bf*)(vb + base + (long)r * 128 + cq);
    VsT[cq + 0][r] = v[0]; VsT[cq + 1][r] = v[1];
    VsT[cq + 2][r] = v[2]; VsT[cq + 3][r] = v[3];
  }
  __syncthreads();

  const int qr = wave * 16 + lx;
  v16bf aq = cat8(*(const v8bf*)&Qs[qr][half * 16],
                  *(const v8bf*)&Qs[qr][half * 16 + 8]);

  float lv[16][8];
#pragma unroll
  for (int kt = 0; kt < 16; ++kt) {
    int kc = kt * 16 + lx;
    v16bf bk = cat8(*(const v8bf*)&Ks[kc][half * 16],
                    *(const v8bf*)&Ks[kc][half * 16 + 8]);
    v8f c0 = (v8f){0,0,0,0,0,0,0,0};
    c0 = wmma_bf16(aq, bk, c0);
    float mv = mask_trans ? pmask[kc * 256 + i] : pmask[i * 256 + kc];
    float mterm = 1e9f * (mv - 1.0f);
#pragma unroll
    for (int r = 0; r < 8; ++r) {
      int qrow = q0 + wave * 16 + half * 8 + r;
      lv[kt][r] = c0[r] * 0.17677669529663687f + mterm +
                  tbb[(long)h * 65536 + (long)qrow * 256 + kc];
    }
  }

  float srow[8];
#pragma unroll
  for (int r = 0; r < 8; ++r) {
    float m = lv[0][r];
#pragma unroll
    for (int kt = 1; kt < 16; ++kt) m = fmaxf(m, lv[kt][r]);
    for (int o = 8; o >= 1; o >>= 1) m = fmaxf(m, __shfl_xor(m, o, 16));
    float s = 0.f;
#pragma unroll
    for (int kt = 0; kt < 16; ++kt) { float e = __expf(lv[kt][r] - m); lv[kt][r] = e; s += e; }
    for (int o = 8; o >= 1; o >>= 1) s += __shfl_xor(s, o, 16);
    srow[r] = s;
  }

  v8f oacc0 = (v8f){0,0,0,0,0,0,0,0};
  v8f oacc1 = (v8f){0,0,0,0,0,0,0,0};
#pragma unroll
  for (int kp = 0; kp < 8; ++kp) {
    // stage attn 16x32 tile (C layout -> LDS [row][k])
#pragma unroll
    for (int t = 0; t < 2; ++t) {
      int kt = kp * 2 + t;
#pragma unroll
      for (int r = 0; r < 8; ++r)
        Aw[wave][half * 8 + r][t * 16 + lx] = (bf16)lv[kt][r];
    }
    asm volatile("s_wait_dscnt 0x0" ::: "memory");  // same-wave LDS RAW (in-order anyway)
    v16bf aa = cat8(*(const v8bf*)&Aw[wave][lx][half * 16],
                    *(const v8bf*)&Aw[wave][lx][half * 16 + 8]);
    {
      int cc = lx;  // ct = 0
      v16bf bv = cat8(*(const v8bf*)&VsT[cc][kp * 32 + half * 16],
                      *(const v8bf*)&VsT[cc][kp * 32 + half * 16 + 8]);
      oacc0 = wmma_bf16(aa, bv, oacc0);
    }
    {
      int cc = 16 + lx;  // ct = 1
      v16bf bv = cat8(*(const v8bf*)&VsT[cc][kp * 32 + half * 16],
                      *(const v8bf*)&VsT[cc][kp * 32 + half * 16 + 8]);
      oacc1 = wmma_bf16(aa, bv, oacc1);
    }
  }

#pragma unroll
  for (int r = 0; r < 8; ++r) {
    int qrow = q0 + wave * 16 + half * 8 + r;
    float inv = 1.f / srow[r];
    long i0 = base + (long)qrow * 128 + lx;
    ob[i0]      = (bf16)(oacc0[r] * inv * gb[i0]);
    ob[i0 + 16] = (bf16)(oacc1[r] * inv * gb[i0 + 16]);
  }
}

// ---------------------------------------------------------------------------
// Host orchestration
// ---------------------------------------------------------------------------
// Input flattening: top-level setup_inputs() order (m, z, msa_mask, pair_mask,
// params), params flattened recursively in sorted-key order (jax pytree).
enum {
  IN_M = 0, IN_Z, IN_MSAMASK, IN_PAIRMASK,
  MSA_L1_B = 4, MSA_L1_W, MSA_L2_B, MSA_L2_W, MSA_LN_B, MSA_LN_G,
  OPM_A_B = 10, OPM_A_W, OPM_B_B, OPM_B_W, OPM_LN_B, OPM_LN_G, OPM_O_B, OPM_O_W,
  PT_L1_B = 18, PT_L1_W, PT_L2_B, PT_L2_W, PT_LN_B, PT_LN_G,
  TAE = 24,   // g.b,g.w,k.w,ln.b,ln.g,o.b,o.w,q.w,tb.w,v.w
  TAS = 34,
  TMI = 44,   // a_g.b,a_g.w,a_p.b,a_p.w,b_g.b,b_g.w,b_p.b,b_p.w,g.b,g.w,ln_in.b,ln_in.g,ln_out.b,ln_out.g,p.b,p.w
  TMO = 60,
};

#define GEMM(EPI, AG, NSUB, P, BATCH) \
  gemm_k<EPI, AG, NSUB><<<dim3((P).M / 128, (P).N / ((NSUB) * 16), (BATCH)), 256, 0, stream>>>(P)

extern "C" void kernel_launch(void* const* d_in, const int* in_sizes, int n_in,
                              void* d_out, int out_size, void* d_ws, size_t ws_size,
                              hipStream_t stream) {
  (void)in_sizes; (void)n_in; (void)out_size;
  const float* m_in  = (const float*)d_in[IN_M];
  const float* z_in  = (const float*)d_in[IN_Z];
  const float* msam  = (const float*)d_in[IN_MSAMASK];
  const float* pairm = (const float*)d_in[IN_PAIRMASK];
  float* mout = (float*)d_out;
  float* zout = mout + 8388608;
  auto F = [&](int i) { return (const float*)d_in[i]; };

  // ---- workspace layout ----
  char* ws = (char*)d_ws; size_t cur = 0;
  auto alloc = [&](size_t bytes) -> char* {
    char* p = ws + cur; cur += (bytes + 255) & ~(size_t)255; return p;
  };
  char*  wtreg = alloc(4u << 20); size_t wtcur = 0;
  bf16*  LNB = (bf16*)alloc(16777216);            // LN output (bf16), up to 32768x256
  char*  BIG = alloc(67108864);                   // MSA hidden / pair hidden / tri-mul bufs
  bf16*  H    = (bf16*)BIG;
  float* XBUF = (float*)BIG;                      // tri-mul X (65536x128 f32)
  bf16*  AT   = (bf16*)(BIG + 33554432);          // tri-mul aT [128][256][256]
  bf16*  BT   = (bf16*)(BIG + 50331648);          // tri-mul bT
  bf16*  C1  = (bf16*)alloc(16777216);            // a / q
  bf16*  C2  = (bf16*)alloc(16777216);            // b / k
  char*  C3r = alloc(33554432);
  float* PBUF = (float*)C3r;                      // tri-mul pre-gate (f32)
  bf16*  VBUF = (bf16*)C3r;                       // attention v (bf16)
  float* GBUF = (float*)alloc(33554432);          // sigmoid gates (f32)
  bf16*  C5  = (bf16*)alloc(16777216);            // OPM chunk / attention out
  float* NORM = (float*)alloc(262144);
  float* TBB  = (float*)alloc(1048576);
  if (cur > ws_size) return;                      // insufficient scratch

  // ---- weights -> bf16 ----
  auto cvt = [&](int idx, size_t n) -> bf16* {
    bf16* dst = (bf16*)(wtreg + wtcur);
    wtcur += (n * 2 + 255) & ~(size_t)255;
    f2bf_k<<<dim3((unsigned)((n + 255) / 256)), 256, 0, stream>>>(F(idx), dst, (int)n);
    return dst;
  };
  bf16* Wm1 = cvt(MSA_L1_W, 262144);
  bf16* Wm2 = cvt(MSA_L2_W, 262144);
  bf16* Woa = cvt(OPM_A_W, 8192);
  bf16* Wob = cvt(OPM_B_W, 8192);
  bf16* Woo = cvt(OPM_O_W, 131072);
  bf16* Wp1 = cvt(PT_L1_W, 65536);
  bf16* Wp2 = cvt(PT_L2_W, 65536);
  int attbase[2] = {TAS, TAE};
  bf16 *Wq[2], *Wk[2], *Wv[2], *Wga[2], *Woat[2];
  for (int a = 0; a < 2; ++a) {
    int B0 = attbase[a];
    Wga[a] = cvt(B0 + 1, 16384); Wk[a] = cvt(B0 + 2, 16384);
    Woat[a] = cvt(B0 + 6, 16384); Wq[a] = cvt(B0 + 7, 16384); Wv[a] = cvt(B0 + 9, 16384);
  }
  int tmbase[2] = {TMO, TMI};
  bf16 *Wag[2], *Wap[2], *Wbg[2], *Wbp[2], *Wgt[2], *Wpt[2];
  for (int t = 0; t < 2; ++t) {
    int B0 = tmbase[t];
    Wag[t] = cvt(B0 + 1, 16384); Wap[t] = cvt(B0 + 3, 16384);
    Wbg[t] = cvt(B0 + 5, 16384); Wbp[t] = cvt(B0 + 7, 16384);
    Wgt[t] = cvt(B0 + 9, 16384); Wpt[t] = cvt(B0 + 15, 16384);
  }

  hipMemcpyAsync(zout, z_in, 33554432, hipMemcpyDeviceToDevice, stream);

  auto ln = [&](const float* in, bf16* out, int gi, int bi, int W, int rows, int tr) {
    ln_k<<<dim3(rows / 8), 256, 0, stream>>>(in, out, F(gi), F(bi), W, rows, tr);
  };

  // ================= 1. MSA transition =================
  ln(m_in, LNB, MSA_LN_G, MSA_LN_B, 256, 32768, 0);
  { GemmP p{}; p.A = LNB; p.B = Wm1; p.bias = F(MSA_L1_B); p.out = H;
    p.M = 32768; p.N = 1024; p.K = 256; p.lda = 256; p.ldb = 1024; p.ldo = 1024;
    GEMM(EPI_RELU_BF16, AG_DIR, 4, p, 1); }
  { GemmP p{}; p.A = H; p.B = Wm2; p.bias = F(MSA_L2_B); p.res = m_in; p.rowmask = msam;
    p.out = mout; p.M = 32768; p.N = 256; p.K = 1024; p.lda = 1024; p.ldb = 256; p.ldo = 256;
    GEMM(EPI_RES_MASK_F32, AG_DIR, 4, p, 1); }

  // ================= 2. Outer product mean =================
  ln(mout, LNB, OPM_LN_G, OPM_LN_B, 256, 32768, 0);
  { GemmP p{}; p.A = LNB; p.B = Woa; p.bias = F(OPM_A_B); p.rowmask = msam; p.out = C1;
    p.M = 32768; p.N = 32; p.K = 256; p.lda = 256; p.ldb = 32; p.ldo = 32;
    GEMM(EPI_MASK_BF16, AG_DIR, 2, p, 1); }
  { GemmP p{}; p.A = LNB; p.B = Wob; p.bias = F(OPM_B_B); p.rowmask = msam; p.out = C2;
    p.M = 32768; p.N = 32; p.K = 256; p.lda = 256; p.ldb = 32; p.ldo = 32;
    GEMM(EPI_MASK_BF16, AG_DIR, 2, p, 1); }
  opmnorm_k<<<256, 256, 0, stream>>>(msam, NORM);
  for (int ch = 0; ch < 8; ++ch) {               // chunk 32 rows of i
    { GemmP p{}; p.A = C1 + ch * 1024; p.B = C2; p.out = C5;   // O1[(di,c),(j,d)]
      p.M = 1024; p.N = 8192; p.K = 128; p.lda = 8192; p.ldb = 8192; p.ldo = 8192;
      GEMM(EPI_BF16, AG_TRANS, 4, p, 1); }
    { GemmP p{}; p.A = C5; p.B = Woo; p.bias = F(OPM_O_B);
      p.norm = NORM + ch * 8192; p.out = zout + (size_t)ch * 8192 * 128;
      p.M = 8192; p.N = 128; p.K = 1024; p.lda = 8192; p.ldb = 128; p.ldo = 128;
      GEMM(EPI_RES_DIV_F32, AG_OPM2, 4, p, 1); }
  }

  // ================= 3/4. Triangle multiplication (out, in) =================
  for (int t = 0; t < 2; ++t) {
    int B0 = tmbase[t];
    int swapA = (t == 0) ? 0 : 1, swapB = (t == 0) ? 1 : 0;
    ln(zout, LNB, B0 + 11, B0 + 10, 128, 65536, 0);          // ln_in
    { GemmP p{}; p.A = LNB; p.B = Wap[t]; p.bias = F(B0 + 2); p.out = PBUF;
      p.M = 65536; p.N = 128; p.K = 128; p.lda = 128; p.ldb = 128; p.ldo = 128;
      GEMM(EPI_F32, AG_DIR, 4, p, 1); }
    { GemmP p{}; p.A = LNB; p.B = Wag[t]; p.bias = F(B0 + 0); p.aux = PBUF;
      p.rowmask = pairm; p.out = C1;
      p.M = 65536; p.N = 128; p.K = 128; p.lda = 128; p.ldb = 128; p.ldo = 128;
      GEMM(EPI_GATE_BF16, AG_DIR, 4, p, 1); }
    { GemmP p{}; p.A = LNB; p.B = Wbp[t]; p.bias = F(B0 + 6); p.out = PBUF;
      p.M = 65536; p.N = 128; p.K = 128; p.lda = 128; p.ldb = 128; p.ldo = 128;
      GEMM(EPI_F32, AG_DIR, 4, p, 1); }
    { GemmP p{}; p.A = LNB; p.B = Wbg[t]; p.bias = F(B0 + 4); p.aux = PBUF;
      p.rowmask = pairm; p.out = C2;
      p.M = 65536; p.N = 128; p.K = 128; p.lda = 128; p.ldb = 128; p.ldo = 128;
      GEMM(EPI_GATE_BF16, AG_DIR, 4, p, 1); }
    { GemmP p{}; p.A = LNB; p.B = Wgt[t]; p.bias = F(B0 + 8); p.out = GBUF;
      p.M = 65536; p.N = 128; p.K = 128; p.lda = 128; p.ldb = 128; p.ldo = 128;
      GEMM(EPI_SIG_F32, AG_DIR, 4, p, 1); }
    trans_cik_k<<<32768, 256, 0, stream>>>(C1, AT, swapA);
    trans_cik_k<<<32768, 256, 0, stream>>>(C2, BT, swapB);
    { GemmP p{}; p.A = AT; p.B = BT; p.out = XBUF;            // batched over 128 channels
      p.M = 256; p.N = 256; p.K = 256; p.lda = 256; p.ldb = 256; p.ldo = 256;
      p.strideA = 65536; p.strideB = 65536;
      GEMM(EPI_SCAT_CH_F32, AG_DIR, 4, p, 128); }
    ln(XBUF, LNB, B0 + 13, B0 + 12, 128, 65536, 0);           // ln_out
    { GemmP p{}; p.A = LNB; p.B = Wpt[t]; p.bias = F(B0 + 14); p.aux = GBUF; p.out = zout;
      p.M = 65536; p.N = 128; p.K = 128; p.lda = 128; p.ldb = 128; p.ldo = 128;
      GEMM(EPI_RES_AUX_F32, AG_DIR, 4, p, 1); }
  }

  // ================= 5/6. Triangle attention (start, end) =================
  for (int a = 0; a < 2; ++a) {
    int B0 = attbase[a]; int tr = a;                 // a=1 -> work on z^T
    ln(zout, LNB, B0 + 4, B0 + 3, 128, 65536, tr);
    { GemmP p{}; p.A = LNB; p.B = Wq[a]; p.out = C1;
      p.M = 65536; p.N = 128; p.K = 128; p.lda = 128; p.ldb = 128; p.ldo = 128;
      GEMM(EPI_BF16, AG_DIR, 4, p, 1); }
    { GemmP p{}; p.A = LNB; p.B = Wk[a]; p.out = C2;
      p.M = 65536; p.N = 128; p.K = 128; p.lda = 128; p.ldb = 128; p.ldo = 128;
      GEMM(EPI_BF16, AG_DIR, 4, p, 1); }
    { GemmP p{}; p.A = LNB; p.B = Wv[a]; p.out = VBUF;
      p.M = 65536; p.N = 128; p.K = 128; p.lda = 128; p.ldb = 128; p.ldo = 128;
      GEMM(EPI_BF16, AG_DIR, 4, p, 1); }
    { GemmP p{}; p.A = LNB; p.B = Wga[a]; p.bias = F(B0 + 0); p.out = GBUF;
      p.M = 65536; p.N = 128; p.K = 128; p.lda = 128; p.ldb = 128; p.ldo = 128;
      GEMM(EPI_SIG_F32, AG_DIR, 4, p, 1); }
    tb_k<<<256, 256, 0, stream>>>(LNB, F(B0 + 8), TBB);
    attn_k<<<dim3(4, 4, 256), 128, 0, stream>>>(C1, C2, VBUF, GBUF, TBB, pairm, C5, tr);
    { GemmP p{}; p.A = C5; p.B = Woat[a]; p.bias = F(B0 + 5); p.out = zout; p.out_trans = tr;
      p.M = 65536; p.N = 128; p.K = 128; p.lda = 128; p.ldb = 128; p.ldo = 128;
      GEMM(EPI_RES_F32, AG_DIR, 4, p, 1); }
  }

  // ================= 7. Pair transition =================
  ln(zout, LNB, PT_LN_G, PT_LN_B, 128, 65536, 0);
  { GemmP p{}; p.A = LNB; p.B = Wp1; p.bias = F(PT_L1_B); p.out = H;
    p.M = 65536; p.N = 512; p.K = 128; p.lda = 128; p.ldb = 512; p.ldo = 512;
    GEMM(EPI_RELU_BF16, AG_DIR, 4, p, 1); }
  { GemmP p{}; p.A = H; p.B = Wp2; p.bias = F(PT_L2_B); p.res = zout; p.rowmask = pairm;
    p.out = zout; p.M = 65536; p.N = 128; p.K = 512; p.lda = 512; p.ldb = 128; p.ldo = 128;
    GEMM(EPI_RES_MASK_F32, AG_DIR, 4, p, 1); }
}
